// PointnetDecoder_75591424409629
// MI455X (gfx1250) — compile-verified, matrix operand order checked by reading, not run on hardware
//
#include <hip/hip_runtime.h>
#include <hip/hip_bf16.h>

// ---------------------------------------------------------------------------
// MI455X (gfx1250) PointNet++-style decoder.
// All MLP/linear layers run through a wave32 WMMA (f16 in / f32 acc) GEMM.
// GEMM hot loop is branch-free: K padded to x32, rows padded to x64,
// weights pre-packed to f16 transposed (Npad x Kpad), LDS fragments loaded
// as 128-bit ds reads matching the ISA 16-bit A-matrix VGPR layout.
// ---------------------------------------------------------------------------

typedef __attribute__((ext_vector_type(16))) _Float16 v16h;
typedef __attribute__((ext_vector_type(8)))  _Float16 v8h;
typedef __attribute__((ext_vector_type(8)))  float    v8f;

#define GEMM_TM 64
#define GEMM_TN 64
#define GEMM_TK 32

// ------------------------- weight pack: W(K,N) f32 -> Wt(Npad,Kpad) f16 ----
__global__ void pack_weight_kernel(const float* __restrict__ W, int K, int N,
                                   int Kpad, int Npad, _Float16* __restrict__ Wt)
{
    int i = blockIdx.x * blockDim.x + threadIdx.x;
    if (i >= Npad * Kpad) return;
    int n = i / Kpad, k = i % Kpad;
    float v = (n < N && k < K) ? W[(size_t)k * N + n] : 0.0f;
    Wt[i] = (_Float16)v;
}

// ------------------------- WMMA GEMM: Y = act(X @ W + b) -------------------
// X: (rows_pad x ldx) f32, ldx % 32 == 0, zero-padded K columns.
// Wt: (Npad x Kpad) f16, transposed + padded.   Y: Mrows x N f32.
__global__ __launch_bounds__(256)
void gemm_bias_act(const float* __restrict__ X, int ldx,
                   const _Float16* __restrict__ Wt, int Kpad,
                   const float* __restrict__ bias, float* __restrict__ Y,
                   int Mrows, int N, int relu)
{
    __shared__ _Float16 As[GEMM_TM * GEMM_TK];   // [m][k]
    __shared__ _Float16 Bs[GEMM_TN * GEMM_TK];   // [n][k]

    const int tid  = threadIdx.x;
    const int lane = tid & 31;
    const int wave = tid >> 5;            // 0..7
    const int m0   = (int)blockIdx.y * GEMM_TM;
    const int n0   = (int)blockIdx.x * GEMM_TN;
    const int wm   = (wave >> 1) * 16;    // 0,16,32,48
    const int wn   = (wave & 1) * 32;     // 0,32

    // staging coordinates: 4 threads per row, 8 elements each
    const int sr = tid >> 2;              // 0..63
    const int sc = (tid & 3) * 8;         // 0,8,16,24

    v8f acc0 = {};
    v8f acc1 = {};

    for (int k0 = 0; k0 < Kpad; k0 += GEMM_TK) {
        // --- A tile 64x32: two float4 global loads, cvt to f16 ---
        {
            const float4* src = (const float4*)(X + (size_t)(m0 + sr) * ldx + k0 + sc);
            float4 v0 = src[0];
            float4 v1 = src[1];
            _Float16* dst = As + sr * GEMM_TK + sc;
            dst[0] = (_Float16)v0.x; dst[1] = (_Float16)v0.y;
            dst[2] = (_Float16)v0.z; dst[3] = (_Float16)v0.w;
            dst[4] = (_Float16)v1.x; dst[5] = (_Float16)v1.y;
            dst[6] = (_Float16)v1.z; dst[7] = (_Float16)v1.w;
        }
        // --- B tile 64x32 (N-major): one 16B global load -> 16B LDS store ---
        {
            const v8h* src = (const v8h*)(Wt + (size_t)(n0 + sr) * Kpad + k0 + sc);
            *(v8h*)(Bs + sr * GEMM_TK + sc) = *src;
        }
        __syncthreads();

        // --- fragments: two contiguous 128-bit LDS loads each ---
        const int kh = (lane >> 4) * 8;
        const _Float16* ap = As + (wm + (lane & 15)) * GEMM_TK;
        v8h alo = *(const v8h*)(ap + kh);
        v8h ahi = *(const v8h*)(ap + 16 + kh);
        v16h a = __builtin_shufflevector(alo, ahi, 0, 1, 2, 3, 4, 5, 6, 7,
                                                   8, 9, 10, 11, 12, 13, 14, 15);

        const _Float16* bp0 = Bs + (wn + (lane & 15)) * GEMM_TK;
        v8h b0lo = *(const v8h*)(bp0 + kh);
        v8h b0hi = *(const v8h*)(bp0 + 16 + kh);
        v16h b0 = __builtin_shufflevector(b0lo, b0hi, 0, 1, 2, 3, 4, 5, 6, 7,
                                                      8, 9, 10, 11, 12, 13, 14, 15);

        const _Float16* bp1 = bp0 + 16 * GEMM_TK;
        v8h b1lo = *(const v8h*)(bp1 + kh);
        v8h b1hi = *(const v8h*)(bp1 + 16 + kh);
        v16h b1 = __builtin_shufflevector(b1lo, b1hi, 0, 1, 2, 3, 4, 5, 6, 7,
                                                      8, 9, 10, 11, 12, 13, 14, 15);

        acc0 = __builtin_amdgcn_wmma_f32_16x16x32_f16(false, a, false, b0, (short)0, acc0, false, false);
        acc1 = __builtin_amdgcn_wmma_f32_16x16x32_f16(false, a, false, b1, (short)0, acc1, false, false);
        __syncthreads();
    }

    // --- epilogue: ISA 16x16 f32 C/D layout, guarded stores ---
    const int colA  = n0 + wn + (lane & 15);
    const int colB  = colA + 16;
    const int rbase = m0 + wm + (lane >> 4) * 8;
#pragma unroll
    for (int i = 0; i < 8; ++i) {
        int gr = rbase + i;
        if (gr >= Mrows) continue;
        if (colA < N) {
            float v = acc0[i] + bias[colA];
            if (relu) v = v > 0.0f ? v : 0.0f;
            Y[(size_t)gr * N + colA] = v;
        }
        if (colB < N) {
            float v = acc1[i] + bias[colB];
            if (relu) v = v > 0.0f ? v : 0.0f;
            Y[(size_t)gr * N + colB] = v;
        }
    }
}

// ------------------------- masked BatchNorm stats/apply --------------------
__global__ __launch_bounds__(256)
void bn_stats_kernel(const float* __restrict__ X, const int* __restrict__ valid,
                     int rows, int C, float* __restrict__ mean, float* __restrict__ rstd)
{
    __shared__ float ssum[256], ssq[256], scnt[256];
    const int c = blockIdx.x;
    float sum = 0.f, sq = 0.f, cnt = 0.f;
    for (int r = threadIdx.x; r < rows; r += 256) {
        float w = valid ? (valid[r] ? 1.f : 0.f) : 1.f;
        float v = X[(size_t)r * C + c] * w;
        sum += v; sq += v * v; cnt += w;
    }
    ssum[threadIdx.x] = sum; ssq[threadIdx.x] = sq; scnt[threadIdx.x] = cnt;
    __syncthreads();
    for (int off = 128; off > 0; off >>= 1) {
        if (threadIdx.x < off) {
            ssum[threadIdx.x] += ssum[threadIdx.x + off];
            ssq[threadIdx.x]  += ssq[threadIdx.x + off];
            scnt[threadIdx.x] += scnt[threadIdx.x + off];
        }
        __syncthreads();
    }
    if (threadIdx.x == 0) {
        float n   = fmaxf(scnt[0], 1.f);
        float mu  = ssum[0] / n;
        float var = fmaxf(ssq[0] / n - mu * mu, 0.f);
        mean[c] = mu;
        rstd[c] = rsqrtf(var + 1e-5f);
    }
}

__global__ void bn_apply_kernel(float* __restrict__ X, int total, int C,
                                const float* __restrict__ mean, const float* __restrict__ rstd,
                                const float* __restrict__ g, const float* __restrict__ bt)
{
    int i = blockIdx.x * blockDim.x + threadIdx.x;
    if (i >= total) return;
    int c = i % C;
    X[i] = (X[i] - mean[c]) * rstd[c] * g[c] + bt[c];
}

// ------------------------- trilinear grid sample ---------------------------
__global__ void grid_sample_kernel(const float* __restrict__ grid,  // (B,C,G,G,G)
                                   const float* __restrict__ qp,    // (B,M,3)
                                   float* __restrict__ x0,          // (B,M,C)
                                   int Bn, int Mn, int Cn, int Gn)
{
    int bm = blockIdx.x;
    int b = bm / Mn, m = bm % Mn;
    const float* p = qp + ((size_t)b * Mn + m) * 3;
    float px = 2.f * p[0] - 1.f, py = 2.f * p[1] - 1.f, pz = 2.f * p[2] - 1.f;
    float gm1 = (float)(Gn - 1);
    float x = fminf(fmaxf((px + 1.f) * 0.5f * gm1, 0.f), gm1);
    float y = fminf(fmaxf((py + 1.f) * 0.5f * gm1, 0.f), gm1);
    float z = fminf(fmaxf((pz + 1.f) * 0.5f * gm1, 0.f), gm1);
    int x0i = (int)floorf(x), y0i = (int)floorf(y), z0i = (int)floorf(z);
    int x1i = min(x0i + 1, Gn - 1), y1i = min(y0i + 1, Gn - 1), z1i = min(z0i + 1, Gn - 1);
    float fx = x - x0i, fy = y - y0i, fz = z - z0i;
    int GG = Gn * Gn;
    size_t gb = (size_t)b * Cn * Gn * GG;
    for (int c = threadIdx.x; c < Cn; c += blockDim.x) {
        const float* gp = grid + gb + (size_t)c * Gn * GG;
        float v =
            gp[z0i * GG + y0i * Gn + x0i] * (1 - fz) * (1 - fy) * (1 - fx) +
            gp[z0i * GG + y0i * Gn + x1i] * (1 - fz) * (1 - fy) * fx +
            gp[z0i * GG + y1i * Gn + x0i] * (1 - fz) * fy * (1 - fx) +
            gp[z0i * GG + y1i * Gn + x1i] * (1 - fz) * fy * fx +
            gp[z1i * GG + y0i * Gn + x0i] * fz * (1 - fy) * (1 - fx) +
            gp[z1i * GG + y0i * Gn + x1i] * fz * (1 - fy) * fx +
            gp[z1i * GG + y1i * Gn + x0i] * fz * fy * (1 - fx) +
            gp[z1i * GG + y1i * Gn + x1i] * fz * fy * fx;
        x0[((size_t)b * Mn + m) * Cn + c] = v;
    }
}

// ------------------------- farthest point sampling -------------------------
__global__ __launch_bounds__(256)
void fps_kernel(const float* __restrict__ pos, int P, int S,
                int* __restrict__ idx, float* __restrict__ mind)
{
    const int b = blockIdx.x;
    const float* p = pos + (size_t)b * P * 3;
    float* md = mind + (size_t)b * P;
    int* id = idx + (size_t)b * S;
    __shared__ float smax[256];
    __shared__ int   simax[256];
    __shared__ int   s_cur;
    const int tid = threadIdx.x;

    float p0x = p[0], p0y = p[1], p0z = p[2];
    for (int i = tid; i < P; i += 256) {
        float dx = p[3 * i] - p0x, dy = p[3 * i + 1] - p0y, dz = p[3 * i + 2] - p0z;
        md[i] = dx * dx + dy * dy + dz * dz;
    }
    if (tid == 0) id[0] = 0;
    __syncthreads();

    for (int s = 1; s < S; ++s) {
        float best = -1.f; int bi = 0x7fffffff;
        for (int i = tid; i < P; i += 256) {
            float v = md[i];
            if (v > best) { best = v; bi = i; }
        }
        smax[tid] = best; simax[tid] = bi;
        __syncthreads();
        for (int off = 128; off > 0; off >>= 1) {
            if (tid < off) {
                bool take = (smax[tid + off] > smax[tid]) ||
                            (smax[tid + off] == smax[tid] && simax[tid + off] < simax[tid]);
                if (take) { smax[tid] = smax[tid + off]; simax[tid] = simax[tid + off]; }
            }
            __syncthreads();
        }
        if (tid == 0) { s_cur = simax[0]; id[s] = s_cur; }
        __syncthreads();
        int nxt = s_cur;
        float nx = p[3 * nxt], ny = p[3 * nxt + 1], nz = p[3 * nxt + 2];
        for (int i = tid; i < P; i += 256) {
            float dx = p[3 * i] - nx, dy = p[3 * i + 1] - ny, dz = p[3 * i + 2] - nz;
            md[i] = fminf(md[i], dx * dx + dy * dy + dz * dz);
        }
        __syncthreads();
    }
}

__global__ void gather_pos_kernel(const float* __restrict__ pos, const int* __restrict__ idx,
                                  int S, int P, float* __restrict__ out, int total)
{
    int t = blockIdx.x * blockDim.x + threadIdx.x;
    if (t >= total) return;
    int b = t / S;
    int j = idx[t];
    out[(size_t)t * 3 + 0] = pos[((size_t)b * P + j) * 3 + 0];
    out[(size_t)t * 3 + 1] = pos[((size_t)b * P + j) * 3 + 1];
    out[(size_t)t * 3 + 2] = pos[((size_t)b * P + j) * 3 + 2];
}

// ------------------------- radius neighborhood -----------------------------
__global__ void radius_kernel(const float* __restrict__ src, const float* __restrict__ qry,
                              int P, int S, float r2, int K,
                              int* __restrict__ nbr, int* __restrict__ valid, int total)
{
    int t = blockIdx.x * blockDim.x + threadIdx.x;
    if (t >= total) return;
    int b = t / S;
    const float* q  = qry + (size_t)t * 3;
    const float* sp = src + (size_t)b * P * 3;
    int* nb = nbr + (size_t)t * K;
    int* vl = valid + (size_t)t * K;
    float qx = q[0], qy = q[1], qz = q[2];
    int cnt = 0;
    for (int j = 0; j < P && cnt < K; ++j) {
        float dx = sp[3 * j] - qx, dy = sp[3 * j + 1] - qy, dz = sp[3 * j + 2] - qz;
        if (dx * dx + dy * dy + dz * dz <= r2) { nb[cnt] = j; vl[cnt] = 1; cnt++; }
    }
    for (int j = 0; j < P && cnt < K; ++j) {
        float dx = sp[3 * j] - qx, dy = sp[3 * j + 1] - qy, dz = sp[3 * j + 2] - qz;
        if (dx * dx + dy * dy + dz * dz > r2) { nb[cnt] = j; vl[cnt] = 0; cnt++; }
    }
}

// ---------------- gather [x[nbr], pos[nbr]-qpos, 0...pad] ------------------
__global__ void gather_sa_feat(const float* __restrict__ x, const float* __restrict__ pos,
                               const float* __restrict__ qpos, const int* __restrict__ nbr,
                               int P, int S, int K, int C, int ldF, float* __restrict__ F)
{
    int row = blockIdx.x;                 // (b*S+s)*K + k
    int bs = row / K;
    int b = bs / S, s = bs % S;
    int j = nbr[row];
    const float* xs = x + ((size_t)b * P + j) * C;
    float* fr = F + (size_t)row * ldF;
    for (int c = threadIdx.x; c < ldF; c += blockDim.x) {
        float v;
        if (c < C)           v = xs[c];
        else if (c < C + 3)  v = pos[((size_t)b * P + j) * 3 + (c - C)] -
                                 qpos[((size_t)b * S + s) * 3 + (c - C)];
        else                 v = 0.f;
        fr[c] = v;
    }
}

// ------------------------- masked max-pool over K --------------------------
__global__ void maxpool_kernel(const float* __restrict__ H, const int* __restrict__ valid,
                               int rowsBS, int K, int C, float* __restrict__ out)
{
    int i = blockIdx.x * blockDim.x + threadIdx.x;
    if (i >= rowsBS * C) return;
    int bs = i / C, c = i % C;
    const float* h = H + (size_t)bs * K * C + c;
    const int* vl = valid + (size_t)bs * K;
    float m = -3.4e38f;
    bool any = false;
    for (int k = 0; k < K; ++k)
        if (vl[k]) { m = fmaxf(m, h[(size_t)k * C]); any = true; }
    out[i] = any ? m : 0.f;
}

__global__ void globalmax_kernel(const float* __restrict__ H, int S, int C,
                                 float* __restrict__ out, int total)
{
    int t = blockIdx.x * blockDim.x + threadIdx.x;
    if (t >= total) return;
    int b = t / C, c = t % C;
    float m = -3.4e38f;
    for (int s = 0; s < S; ++s) m = fmaxf(m, H[((size_t)b * S + s) * C + c]);
    out[t] = m;
}

// ------------------------- concat helpers (zero-padded) --------------------
__global__ void concat_rowwise_kernel(const float* __restrict__ A, int Ca,
                                      const float* __restrict__ Bp, int Cb,
                                      int ldF, float* __restrict__ F)
{
    int r = blockIdx.x;
    float* fr = F + (size_t)r * ldF;
    for (int c = threadIdx.x; c < ldF; c += blockDim.x) {
        float v;
        if (c < Ca)           v = A[(size_t)r * Ca + c];
        else if (c < Ca + Cb) v = Bp[(size_t)r * Cb + (c - Ca)];
        else                  v = 0.f;
        fr[c] = v;
    }
}

__global__ void concat_broadcast_kernel(const float* __restrict__ A /*(B,Ca)*/, int Ca,
                                        const float* __restrict__ Bp /*(rows,Cb)*/, int Cb,
                                        int S, int ldF, float* __restrict__ F)
{
    int r = blockIdx.x;
    int b = r / S;
    float* fr = F + (size_t)r * ldF;
    for (int c = threadIdx.x; c < ldF; c += blockDim.x) {
        float v;
        if (c < Ca)           v = A[(size_t)b * Ca + c];
        else if (c < Ca + Cb) v = Bp[(size_t)r * Cb + (c - Ca)];
        else                  v = 0.f;
        fr[c] = v;
    }
}

// ------------------------- kNN (k=3) inverse-distance interp ---------------
__global__ __launch_bounds__(128)
void knn_interp3_kernel(const float* __restrict__ xs, const float* __restrict__ ps,
                        const float* __restrict__ pt, int Ps, int Pt, int C,
                        float* __restrict__ out)
{
    int t = blockIdx.x;
    int b = t / Pt;
    const float* q  = pt + (size_t)t * 3;
    const float* sp = ps + (size_t)b * Ps * 3;
    float qx = q[0], qy = q[1], qz = q[2];

    float d0 = 3.4e38f, d1 = 3.4e38f, d2 = 3.4e38f;
    int   i0 = -1, i1 = -1, i2 = -1;
    for (int j = threadIdx.x; j < Ps; j += 128) {
        float dx = sp[3 * j] - qx, dy = sp[3 * j + 1] - qy, dz = sp[3 * j + 2] - qz;
        float d = dx * dx + dy * dy + dz * dz;
        if (d < d0)      { d2 = d1; i2 = i1; d1 = d0; i1 = i0; d0 = d; i0 = j; }
        else if (d < d1) { d2 = d1; i2 = i1; d1 = d; i1 = j; }
        else if (d < d2) { d2 = d; i2 = j; }
    }
    __shared__ float sd[128 * 3];
    __shared__ int   si[128 * 3];
    sd[threadIdx.x * 3 + 0] = d0; si[threadIdx.x * 3 + 0] = i0;
    sd[threadIdx.x * 3 + 1] = d1; si[threadIdx.x * 3 + 1] = i1;
    sd[threadIdx.x * 3 + 2] = d2; si[threadIdx.x * 3 + 2] = i2;
    __syncthreads();

    __shared__ float w[3], winv;
    __shared__ int   sidx[3];
    if (threadIdx.x == 0) {
        float bd[3] = {3.4e38f, 3.4e38f, 3.4e38f};
        int   bi[3] = {0, 0, 0};
        for (int n = 0; n < 128 * 3; ++n) {
            float d = sd[n]; int j = si[n];
            if (j < 0) continue;
            if (d < bd[0])      { bd[2]=bd[1]; bi[2]=bi[1]; bd[1]=bd[0]; bi[1]=bi[0]; bd[0]=d; bi[0]=j; }
            else if (d < bd[1]) { bd[2]=bd[1]; bi[2]=bi[1]; bd[1]=d; bi[1]=j; }
            else if (d < bd[2]) { bd[2]=d; bi[2]=j; }
        }
        float s = 0.f;
        for (int n = 0; n < 3; ++n) {
            float ww = 1.f / fmaxf(bd[n], 1e-16f);
            w[n] = ww; sidx[n] = bi[n]; s += ww;
        }
        winv = 1.f / s;
    }
    __syncthreads();

    const float* xb = xs + (size_t)b * Ps * C;
    for (int c = threadIdx.x; c < C; c += 128) {
        float v = w[0] * xb[(size_t)sidx[0] * C + c] +
                  w[1] * xb[(size_t)sidx[1] * C + c] +
                  w[2] * xb[(size_t)sidx[2] * C + c];
        out[(size_t)t * C + c] = v * winv;
    }
}

// ===========================================================================
extern "C" void kernel_launch(void* const* d_in, const int* in_sizes, int n_in,
                              void* d_out, int out_size, void* d_ws, size_t ws_size,
                              hipStream_t stream)
{
    (void)in_sizes; (void)n_in; (void)out_size; (void)ws_size;
    const int Bn = 2, Mn = 6000, Cin = 128, Gn = 32, Kn = 64;
    const int S1 = 3000, S2 = 750;
    const float R1 = 0.05f, R2 = 0.1f;

    const float* grid = (const float*)d_in[0];
    const float* qp   = (const float*)d_in[1];
    auto Pm = [&](int i) { return (const float*)d_in[i]; };
    auto ceil32 = [](int x) { return (x + 31) & ~31; };
    auto ceil64 = [](int x) { return (x + 63) & ~63; };

    // --- workspace carve (deterministic bump allocator) ---
    char* base = (char*)d_ws;
    size_t off = 0;
    auto allocF = [&](size_t n) { float*    p = (float*)(base + off);    off += ((n * 4 + 255) / 256) * 256; return p; };
    auto allocI = [&](size_t n) { int*      p = (int*)(base + off);      off += ((n * 4 + 255) / 256) * 256; return p; };
    auto allocH = [&](size_t n) { _Float16* p = (_Float16*)(base + off); off += ((n * 2 + 255) / 256) * 256; return p; };

    float* x0   = allocF((size_t)Bn * Mn * Cin);
    float* mind = allocF((size_t)Bn * Mn);
    int*   idx1 = allocI((size_t)Bn * S1);
    float* pos1 = allocF((size_t)Bn * S1 * 3);
    int*   nbr1 = allocI((size_t)Bn * S1 * Kn);
    int*   val1 = allocI((size_t)Bn * S1 * Kn);
    int*   idx2 = allocI((size_t)Bn * S2);
    float* pos2 = allocF((size_t)Bn * S2 * 3);
    int*   nbr2 = allocI((size_t)Bn * S2 * Kn);
    int*   val2 = allocI((size_t)Bn * S2 * Kn);
    float* x1   = allocF((size_t)Bn * S1 * 128);
    float* x2   = allocF((size_t)Bn * S2 * 256);
    float* x3   = allocF((size_t)Bn * 1024);
    float* x2f  = allocF((size_t)ceil64(Bn * S2) * 256);   // row-padded (GEMM input)
    float* x1f  = allocF((size_t)ceil64(Bn * S1) * 128);   // row-padded
    float* xi2  = allocF((size_t)Bn * S1 * 256);
    float* xi1  = allocF((size_t)Bn * Mn * 128);
    float* x0f  = allocF((size_t)ceil64(Bn * Mn) * 128);   // row-padded
    float* bn_mean = allocF(1024);
    float* bn_rstd = allocF(1024);
    float* bufF = allocF((size_t)Bn * S1 * Kn * 160);      // padded gathered/concat features
    float* bufA = allocF((size_t)Bn * S1 * Kn * 128);      // ping
    float* bufB = allocF((size_t)Bn * S1 * Kn * 128);      // pong

    // pack f32 weight (K,N) -> f16 transposed (Npad,Kpad), then run GEMM
    auto run_gemm = [&](const float* Xin, int ldx, int wi, float* Yout,
                        int rows, int K, int N, int relu) {
        int Kpad = ceil32(K), Npad = ceil64(N);
        _Float16* Wt = allocH((size_t)Npad * Kpad);
        int wtot = Npad * Kpad;
        pack_weight_kernel<<<(wtot + 255) / 256, 256, 0, stream>>>(Pm(wi), K, N, Kpad, Npad, Wt);
        dim3 g((unsigned)((N + 63) / 64), (unsigned)((rows + 63) / 64));
        gemm_bias_act<<<g, 256, 0, stream>>>(Xin, ldx, Wt, Kpad, Pm(wi + 1), Yout, rows, N, relu);
    };
    auto mlp_layer = [&](const float* Xin, int ldx, float* Yout, int rows,
                         int K, int N, int wi, const int* valid) {
        run_gemm(Xin, ldx, wi, Yout, rows, K, N, 1);
        bn_stats_kernel<<<N, 256, 0, stream>>>(Yout, valid, rows, N, bn_mean, bn_rstd);
        int tot = rows * N;
        bn_apply_kernel<<<(tot + 255) / 256, 256, 0, stream>>>(Yout, tot, N, bn_mean, bn_rstd,
                                                               Pm(wi + 2), Pm(wi + 3));
    };

    // 1. trilinear grid sample -> x0 (B,M,128)
    grid_sample_kernel<<<Bn * Mn, 128, 0, stream>>>(grid, qp, x0, Bn, Mn, Cin, Gn);

    // 2. SA1: FPS -> radius -> gather(ld160) -> MLP(131,128,128,128)+BN -> maxpool
    fps_kernel<<<Bn, 256, 0, stream>>>(qp, Mn, S1, idx1, mind);
    gather_pos_kernel<<<(Bn * S1 + 255) / 256, 256, 0, stream>>>(qp, idx1, S1, Mn, pos1, Bn * S1);
    radius_kernel<<<(Bn * S1 + 127) / 128, 128, 0, stream>>>(qp, pos1, Mn, S1, R1 * R1, Kn, nbr1, val1, Bn * S1);
    gather_sa_feat<<<Bn * S1 * Kn, 128, 0, stream>>>(x0, qp, pos1, nbr1, Mn, S1, Kn, Cin, 160, bufF);
    {
        int rows = Bn * S1 * Kn;   // 384000, multiple of 64
        mlp_layer(bufF, 160, bufA, rows, 131, 128, 2, val1);
        mlp_layer(bufA, 128, bufB, rows, 128, 128, 6, val1);
        mlp_layer(bufB, 128, bufA, rows, 128, 128, 10, val1);
        maxpool_kernel<<<(Bn * S1 * 128 + 255) / 256, 256, 0, stream>>>(bufA, val1, Bn * S1, Kn, 128, x1);
    }

    // 3. SA2: MLP(131,128,128,256)
    fps_kernel<<<Bn, 256, 0, stream>>>(pos1, S1, S2, idx2, mind);
    gather_pos_kernel<<<(Bn * S2 + 255) / 256, 256, 0, stream>>>(pos1, idx2, S2, S1, pos2, Bn * S2);
    radius_kernel<<<(Bn * S2 + 127) / 128, 128, 0, stream>>>(pos1, pos2, S1, S2, R2 * R2, Kn, nbr2, val2, Bn * S2);
    gather_sa_feat<<<Bn * S2 * Kn, 128, 0, stream>>>(x1, pos1, pos2, nbr2, S1, S2, Kn, 128, 160, bufF);
    {
        int rows = Bn * S2 * Kn;   // 96000, multiple of 64
        mlp_layer(bufF, 160, bufA, rows, 131, 128, 14, val2);
        mlp_layer(bufA, 128, bufB, rows, 128, 128, 18, val2);
        mlp_layer(bufB, 128, bufA, rows, 128, 256, 22, val2);
        maxpool_kernel<<<(Bn * S2 * 256 + 255) / 256, 256, 0, stream>>>(bufA, val2, Bn * S2, Kn, 256, x2);
    }

    // 4. SA3: MLP(259->288pad,256,512,1024) + global max -> x3 (B,1024)
    concat_rowwise_kernel<<<Bn * S2, 128, 0, stream>>>(x2, 256, pos2, 3, 288, bufF);
    {
        int rows = Bn * S2;        // 1500 (store-guarded; buffers row-padded)
        mlp_layer(bufF, 288, bufA, rows, 259, 256, 26, nullptr);
        mlp_layer(bufA, 256, bufB, rows, 256, 512, 30, nullptr);
        mlp_layer(bufB, 512, bufA, rows, 512, 1024, 34, nullptr);
        globalmax_kernel<<<(Bn * 1024 + 255) / 256, 256, 0, stream>>>(bufA, S2, 1024, x3, Bn * 1024);
    }

    // 5. FP3: broadcast x3 + x2 -> MLP(1280,256,256) -> x2f
    concat_broadcast_kernel<<<Bn * S2, 128, 0, stream>>>(x3, 1024, x2, 256, S2, 1280, bufF);
    {
        int rows = Bn * S2;
        mlp_layer(bufF, 1280, bufA, rows, 1280, 256, 38, nullptr);
        mlp_layer(bufA, 256, x2f, rows, 256, 256, 42, nullptr);
    }

    // 6. FP2: kNN3 interp (pos2->pos1) + x1 -> MLP(384,256,128) -> x1f
    knn_interp3_kernel<<<Bn * S1, 128, 0, stream>>>(x2f, pos2, pos1, S2, S1, 256, xi2);
    concat_rowwise_kernel<<<Bn * S1, 128, 0, stream>>>(xi2, 256, x1, 128, 384, bufF);
    mlp_layer(bufF, 384, bufA, Bn * S1, 384, 256, 46, nullptr);
    mlp_layer(bufA, 256, x1f, Bn * S1, 256, 128, 50, nullptr);

    // 7. FP1: kNN3 interp (pos1->pos0) + x0 -> MLP(256,128,128,128) -> x0f
    knn_interp3_kernel<<<Bn * Mn, 128, 0, stream>>>(x1f, pos1, qp, S1, Mn, 128, xi1);
    concat_rowwise_kernel<<<Bn * Mn, 128, 0, stream>>>(xi1, 128, x0, 128, 256, bufF);
    mlp_layer(bufF, 256, bufA, Bn * Mn, 256, 128, 54, nullptr);
    mlp_layer(bufA, 128, bufB, Bn * Mn, 128, 128, 58, nullptr);
    mlp_layer(bufB, 128, x0f, Bn * Mn, 128, 128, 62, nullptr);

    // 8. heads: relu(lin1) -> lin2 -> lin3 -> logits (B,1,M,3) f32
    run_gemm(x0f, 128, 66, bufA, Bn * Mn, 128, 128, 1);
    run_gemm(bufA, 128, 68, bufB, Bn * Mn, 128, 128, 0);
    run_gemm(bufB, 128, 70, (float*)d_out, Bn * Mn, 128, 3, 0);
}
